// MultiHeadAttention_40200893890966
// MI455X (gfx1250) — matrix operand order repaired and checked by reading, not compiled, not run on hardware
//
#include <hip/hip_runtime.h>
#include <math.h>

typedef __bf16 bf16;
typedef __attribute__((ext_vector_type(16))) __bf16 v16bf;
typedef __attribute__((ext_vector_type(8)))  float  v8f;

#define WMMA_BF16(a, b, c) \
  __builtin_amdgcn_wmma_f32_16x16x32_bf16(false, (a), false, (b), (short)0, (c), false, false)

// ---------------------------------------------------------------------------
// Fragment loaders: exact gfx1250 wave32 WMMA VGPR layouts (ISA 05_wmma 7.12.2)
// A is 16x32 (MxK) bf16, staged in LDS as [m][k] (k contiguous).
static __device__ __forceinline__ v16bf load_frag_a(const bf16* __restrict__ s, int ld) {
  const int lane = threadIdx.x & 31;
  const int m = lane & 15;
  const int hi = lane >> 4;
  v16bf a;
#pragma unroll
  for (int j = 0; j < 8; ++j) {
    int k0 = ((j < 4) ? (2 * j) : (16 + 2 * (j - 4))) + hi * 8;
    a[2 * j]     = s[m * ld + k0];
    a[2 * j + 1] = s[m * ld + k0 + 1];
  }
  return a;
}

// B is 32x16 (KxN) bf16, staged in LDS transposed as [n][k] (k contiguous).
static __device__ __forceinline__ v16bf load_frag_b(const bf16* __restrict__ s, int ld) {
  const int lane = threadIdx.x & 31;
  const int n = lane & 15;
  const int hi = lane >> 4;
  v16bf b;
#pragma unroll
  for (int j = 0; j < 8; ++j) {
    int k0 = ((j < 4) ? (2 * j) : (16 + 2 * (j - 4))) + hi * 8;
    b[2 * j]     = s[n * ld + k0];
    b[2 * j + 1] = s[n * ld + k0 + 1];
  }
  return b;
}

// ---------------------------------------------------------------------------
// Generic GEMM: C[M,N] = A[M,K] @ W[K,N] + bias (+ optional residual).
// Workgroup tile 128x64, 8 waves; wave w owns rows w*16..w*16+15 x 64 cols.
template <bool A_BF16, bool RESID>
__global__ __launch_bounds__(256) void gemm_kernel(
    const float* __restrict__ Af, const bf16* __restrict__ Ab,
    const float* __restrict__ W, const float* __restrict__ bias,
    const float* __restrict__ resid, bf16* __restrict__ outB,
    float* __restrict__ outF, int M, int N, int K) {
  constexpr int LDA = 34, LDB = 34;
  __shared__ bf16 sA[128 * LDA];
  __shared__ bf16 sB[64 * LDB];
  const int tid = threadIdx.x;
  const int m0 = blockIdx.x * 128;
  const int n0 = blockIdx.y * 64;
  const int w = tid >> 5, lane = tid & 31, hi = lane >> 4, ln = lane & 15;

  v8f acc[4] = {};

  for (int kt = 0; kt < K; kt += 32) {
#pragma unroll
    for (int i = 0; i < 16; ++i) {           // stage A tile 128x32
      int idx = tid + i * 256;
      int r = idx >> 5, c = idx & 31;
      size_t g = (size_t)(m0 + r) * K + kt + c;
      sA[r * LDA + c] = A_BF16 ? Ab[g] : (bf16)Af[g];
    }
#pragma unroll
    for (int i = 0; i < 8; ++i) {            // stage W tile 32x64 transposed
      int idx = tid + i * 256;
      int n = idx & 63, kk = idx >> 6;
      sB[n * LDB + kk] = (bf16)W[(size_t)(kt + kk) * N + n0 + n];
    }
    if (kt + 32 < K)                         // gfx1250 global_prefetch_b8
      __builtin_prefetch(&W[(size_t)(kt + 32 + (tid >> 6)) * N + n0 + (tid & 63)], 0, 1);
    __syncthreads();
    v16bf a = load_frag_a(sA + (w * 16) * LDA, LDA);
#pragma unroll
    for (int t = 0; t < 4; ++t) {
      v16bf b = load_frag_b(sB + (t * 16) * LDB, LDB);
      acc[t] = WMMA_BF16(a, b, acc[t]);
    }
    __syncthreads();
  }

#pragma unroll
  for (int t = 0; t < 4; ++t) {
#pragma unroll
    for (int j = 0; j < 8; ++j) {            // C: VGPR j -> row j (+8 hi), col = lane&15
      int gr = m0 + w * 16 + j + hi * 8;
      int gc = n0 + t * 16 + ln;
      float v = acc[t][j] + bias[gc];
      if (RESID) {
        v += resid[(size_t)gr * N + gc];
        outF[(size_t)gr * N + gc] = v;
      } else {
        outB[(size_t)gr * N + gc] = (bf16)v;
      }
    }
  }
}

// ---------------------------------------------------------------------------
// Fused attention: per 128-query strip of one (b,h):
//   pass 1: stream 64-key tiles, WMMA scores, online softmax stats (m,l) in regs
//   pass 2: recompute scores, p = exp(s-m)/l, NT-store attn (only HBM touch),
//           relayout p via wave-private LDS rows, WMMA-accumulate ctx = p @ v.
// attn HBM traffic: single 400MB write (was write+read+write+read).
__global__ __launch_bounds__(256) void attn_fused_kernel(
    const bf16* __restrict__ qb, const bf16* __restrict__ kb,
    const bf16* __restrict__ vb, const unsigned char* __restrict__ mask,
    float* __restrict__ attn, bf16* __restrict__ ctx) {
  constexpr int LDQ = 66, LDK = 66, LDV = 66, LDP = 66;
  __shared__ bf16 sQ[128 * LDQ];
  __shared__ bf16 sK[64 * LDK];
  __shared__ bf16 sV[64 * LDV];   // [dk][key]
  __shared__ bf16 sP[128 * LDP];
  const int tid = threadIdx.x;
  const int b = blockIdx.y / 12, h = blockIdx.y % 12;
  const int q0 = blockIdx.x * 128;
  const int w = tid >> 5, lane = tid & 31, hi = lane >> 4, ln = lane & 15;

  // stage the Q strip once: 128 x 64
#pragma unroll
  for (int i = 0; i < 32; ++i) {
    int idx = tid + i * 256;
    int r = idx >> 6, c = idx & 63;
    sQ[r * LDQ + c] = qb[((size_t)(b * 1024 + q0 + r)) * 768 + h * 64 + c];
  }
  __syncthreads();

  float m[8], l[8];
#pragma unroll
  for (int j = 0; j < 8; ++j) { m[j] = -3.0e38f; l[j] = 0.f; }

  // ---------------- pass 1: softmax statistics ----------------
  for (int k0 = 0; k0 < 1024; k0 += 64) {
#pragma unroll
    for (int i = 0; i < 16; ++i) {
      int idx = tid + i * 256;
      int r = idx >> 6, c = idx & 63;
      sK[r * LDK + c] = kb[((size_t)(b * 1024 + k0 + r)) * 768 + h * 64 + c];
    }
    __syncthreads();
    v8f acc[4] = {};
#pragma unroll
    for (int ks = 0; ks < 2; ++ks) {
      v16bf a = load_frag_a(sQ + (w * 16) * LDQ + ks * 32, LDQ);
#pragma unroll
      for (int t = 0; t < 4; ++t) {
        v16bf bb = load_frag_b(sK + (t * 16) * LDK + ks * 32, LDK);
        acc[t] = WMMA_BF16(a, bb, acc[t]);
      }
    }
#pragma unroll
    for (int t = 0; t < 4; ++t) {
#pragma unroll
      for (int j = 0; j < 8; ++j) {
        int qg = q0 + w * 16 + j + hi * 8;
        int kg = k0 + t * 16 + ln;
        float s = acc[t][j] * 0.125f;
        if (mask[((size_t)(b * 1024 + qg)) * 1024 + kg]) s = -1e9f;
        acc[t][j] = s;
      }
    }
    // per-row online update; each row lives in one 16-lane half-wave
#pragma unroll
    for (int j = 0; j < 8; ++j) {
      float tmax = fmaxf(fmaxf(acc[0][j], acc[1][j]), fmaxf(acc[2][j], acc[3][j]));
#pragma unroll
      for (int o = 8; o > 0; o >>= 1) tmax = fmaxf(tmax, __shfl_xor(tmax, o, 32));
      float mn = fmaxf(m[j], tmax);
      float ssum = 0.f;
#pragma unroll
      for (int t = 0; t < 4; ++t) ssum += __expf(acc[t][j] - mn);
#pragma unroll
      for (int o = 8; o > 0; o >>= 1) ssum += __shfl_xor(ssum, o, 32);
      l[j] = l[j] * __expf(m[j] - mn) + ssum;
      m[j] = mn;
    }
    __syncthreads();
  }

  float inv[8];
#pragma unroll
  for (int j = 0; j < 8; ++j) inv[j] = 1.0f / l[j];

  // ---------------- pass 2: attn write + ctx accumulate ----------------
  v8f cacc[4] = {};
  for (int k0 = 0; k0 < 1024; k0 += 64) {
#pragma unroll
    for (int i = 0; i < 16; ++i) {
      int idx = tid + i * 256;
      int r = idx >> 6, c = idx & 63;
      sK[r * LDK + c] = kb[((size_t)(b * 1024 + k0 + r)) * 768 + h * 64 + c];
      // V tile transposed to [dk][key]
      sV[c * LDV + r] = vb[((size_t)(b * 1024 + k0 + r)) * 768 + h * 64 + c];
    }
    __syncthreads();
    v8f acc[4] = {};
#pragma unroll
    for (int ks = 0; ks < 2; ++ks) {
      v16bf a = load_frag_a(sQ + (w * 16) * LDQ + ks * 32, LDQ);
#pragma unroll
      for (int t = 0; t < 4; ++t) {
        v16bf bb = load_frag_b(sK + (t * 16) * LDK + ks * 32, LDK);
        acc[t] = WMMA_BF16(a, bb, acc[t]);
      }
    }
    // p = exp(s-m)/l ; NT-store attn; stash p rows in wave-private LDS
#pragma unroll
    for (int t = 0; t < 4; ++t) {
#pragma unroll
      for (int j = 0; j < 8; ++j) {
        int rloc = w * 16 + j + hi * 8;
        int qg = q0 + rloc;
        int kg = k0 + t * 16 + ln;
        float s = acc[t][j] * 0.125f;
        if (mask[((size_t)(b * 1024 + qg)) * 1024 + kg]) s = -1e9f;
        float p = __expf(s - m[j]) * inv[j];
        __builtin_nontemporal_store(
            p, &attn[((size_t)(b * 12 + h) * 1024 + qg) * 1024 + kg]);
        sP[rloc * LDP + t * 16 + ln] = (bf16)p;
      }
    }
    // wave reads back only its own 16 rows -> no cross-wave barrier needed
#pragma unroll
    for (int ks = 0; ks < 2; ++ks) {
      v16bf a = load_frag_a(sP + (w * 16) * LDP + ks * 32, LDP);
#pragma unroll
      for (int t = 0; t < 4; ++t) {
        v16bf bb = load_frag_b(sV + (t * 16) * LDV + ks * 32, LDV);
        cacc[t] = WMMA_BF16(a, bb, cacc[t]);
      }
    }
    __syncthreads();
  }

#pragma unroll
  for (int t = 0; t < 4; ++t) {
#pragma unroll
    for (int j = 0; j < 8; ++j) {
      int qg = q0 + w * 16 + j + hi * 8;
      int dk = t * 16 + ln;
      ctx[((size_t)(b * 1024 + qg)) * 768 + h * 64 + dk] = (bf16)cacc[t][j];
    }
  }
}

// ---------------------------------------------------------------------------
// Rowwise LayerNorm over 768 (= 3 * 256 threads); NT store of final output.
__global__ __launch_bounds__(256) void layernorm_kernel(
    const float* __restrict__ x, const float* __restrict__ gamma,
    const float* __restrict__ beta, float* __restrict__ out) {
  __shared__ float red[8];
  const int tid = threadIdx.x;
  const float* p = x + (size_t)blockIdx.x * 768;
  float v0 = p[tid], v1 = p[tid + 256], v2 = p[tid + 512];
  float s = v0 + v1 + v2;
#pragma unroll
  for (int o = 16; o > 0; o >>= 1) s += __shfl_xor(s, o, 32);
  if ((tid & 31) == 0) red[tid >> 5] = s;
  __syncthreads();
  float tot = 0.f;
#pragma unroll
  for (int i = 0; i < 8; ++i) tot += red[i];
  const float mu = tot * (1.0f / 768.0f);
  float d0 = v0 - mu, d1 = v1 - mu, d2 = v2 - mu;
  float q = d0 * d0 + d1 * d1 + d2 * d2;
#pragma unroll
  for (int o = 16; o > 0; o >>= 1) q += __shfl_xor(q, o, 32);
  __syncthreads();
  if ((tid & 31) == 0) red[tid >> 5] = q;
  __syncthreads();
  float tv = 0.f;
#pragma unroll
  for (int i = 0; i < 8; ++i) tv += red[i];
  const float rs = rsqrtf(tv * (1.0f / 768.0f) + 1e-5f);
  float* o = out + (size_t)blockIdx.x * 768;
  __builtin_nontemporal_store(d0 * rs * gamma[tid]       + beta[tid],       &o[tid]);
  __builtin_nontemporal_store(d1 * rs * gamma[tid + 256] + beta[tid + 256], &o[tid + 256]);
  __builtin_nontemporal_store(d2 * rs * gamma[tid + 512] + beta[tid + 512], &o[tid + 512]);
}

// ---------------------------------------------------------------------------
extern "C" void kernel_launch(void* const* d_in, const int* in_sizes, int n_in,
                              void* d_out, int out_size, void* d_ws, size_t ws_size,
                              hipStream_t stream) {
  const float* Q  = (const float*)d_in[0];
  const float* K  = (const float*)d_in[1];
  const float* V  = (const float*)d_in[2];
  const unsigned char* mask = (const unsigned char*)d_in[3];
  const float* Wq = (const float*)d_in[4];
  const float* bq = (const float*)d_in[5];
  const float* Wk = (const float*)d_in[6];
  const float* bk = (const float*)d_in[7];
  const float* Wv = (const float*)d_in[8];
  const float* bv = (const float*)d_in[9];
  const float* Wo = (const float*)d_in[10];
  const float* bo = (const float*)d_in[11];
  const float* gamma = (const float*)d_in[12];
  const float* beta  = (const float*)d_in[13];

  constexpr size_t MS = (size_t)8192 * 768;   // B*S x D elements
  char* ws = (char*)d_ws;
  bf16* qb  = (bf16*)ws;                      // [8192,768] bf16
  bf16* kb  = qb + MS;
  bf16* vb  = kb + MS;
  bf16* ctx = vb + MS;
  float* outpre = (float*)(ws + 4 * MS * sizeof(bf16));  // [8192,768] fp32

  float* outp  = (float*)d_out;               // out: 6,291,456 fp32
  float* attnp = outp + MS;                   // attn: 100,663,296 fp32

  const dim3 blk(256);
  const dim3 gP(64, 12);                      // 8192/128 x 768/64
  gemm_kernel<false, false><<<gP, blk, 0, stream>>>(Q, nullptr, Wq, bq, nullptr, qb, nullptr, 8192, 768, 768);
  gemm_kernel<false, false><<<gP, blk, 0, stream>>>(K, nullptr, Wk, bk, nullptr, kb, nullptr, 8192, 768, 768);
  gemm_kernel<false, false><<<gP, blk, 0, stream>>>(V, nullptr, Wv, bv, nullptr, vb, nullptr, 8192, 768, 768);
  attn_fused_kernel<<<dim3(8, 96), blk, 0, stream>>>(qb, kb, vb, mask, attnp, ctx);
  gemm_kernel<true, true><<<gP, blk, 0, stream>>>(nullptr, ctx, Wo, bo, Q, nullptr, outpre, 8192, 768, 768);
  layernorm_kernel<<<dim3(8192), blk, 0, stream>>>(outpre, gamma, beta, outp);
}
